// MLPSpeculator_65034394796440
// MI455X (gfx1250) — matrix operand order repaired and checked by reading, not compiled, not run on hardware
//
#include <hip/hip_runtime.h>
#include <cmath>

// ---------------------------------------------------------------------------
// MLPSpeculator for MI455X (gfx1250, wave32, WMMA)
//   3 stages x { proj GEMM + alpha*emb gather  ->  RMSnorm+GELU  ->  head GEMM }
// GEMMs use V_WMMA_F32_16X16X32_BF16 with bf16-split (hi+lo) fp32 emulation:
//   A*B ~= Ah*Bh + Ah*Bl + Al*Bh  (fp32 accumulate)  -- 3 wmma per tile pair.
// LDS pitch is 80B so all fragment traffic is ds_load_b128 / ds_store_b128;
// next K-slab global loads are software-pipelined under the WMMA block.
// ---------------------------------------------------------------------------

typedef __attribute__((ext_vector_type(16))) __bf16        bf16x16;
typedef __attribute__((ext_vector_type(8)))  float         v8f;
typedef __attribute__((ext_vector_type(8)))  unsigned int  u32x8;
typedef __attribute__((ext_vector_type(4)))  unsigned int  u32x4;

#define BM   128
#define BN   128
#define BK   32
#define LDT  40          // LDS row pitch in ushorts (80B = 16B-aligned rows)

#define SEQ_N   512      // sequence positions
#define INDS_LD 515      // N + NPRED
#define INNER   1024
#define VOCAB   32000

__device__ __forceinline__ v8f wmma_bf16(bf16x16 a, bf16x16 b, v8f c) {
    // (neg_a, A, neg_b, B, c_mod, C, reuse_a, reuse_b)
    return __builtin_amdgcn_wmma_f32_16x16x32_bf16(false, a, false, b,
                                                   (short)0, c, false, false);
}

// A fragment, 16x32 bf16 (M x K) per documented CDNA5 layout:
//   lanes 0-15: M=lane,    K groups {0..7, 16..23}
//   lanes16-31: M=lane-16, K groups {8..15, 24..31}
// Each group of 8 consecutive K = 16B -> one ds_load_b128.
__device__ __forceinline__ bf16x16 fragA(const unsigned short* base, int lane, int mbase) {
    const unsigned short* row = base + (mbase + (lane & 15)) * LDT + ((lane >> 4) << 3);
    u32x4 q0 = *(const u32x4*)(row);          // K = kb .. kb+7
    u32x4 q1 = *(const u32x4*)(row + 16);     // K = kb+16 .. kb+23
    return __builtin_bit_cast(bf16x16,
        __builtin_shufflevector(q0, q1, 0, 1, 2, 3, 4, 5, 6, 7));
}

// B fragment, 32x16 bf16 (K x N): lanes 0-15 carry column N=lane, K {0..7, 16..23};
// lanes 16-31 carry K {8..15, 24..31}. Each group of 8 K = 16B -> one ds_load_b128.
__device__ __forceinline__ bf16x16 fragB(const unsigned short* base, int lane, int nbase) {
    const unsigned short* row = base + (nbase + (lane & 15)) * LDT + ((lane >> 4) << 3);
    u32x4 q0 = *(const u32x4*)(row);
    u32x4 q1 = *(const u32x4*)(row + 16);
    return __builtin_bit_cast(bf16x16,
        __builtin_shufflevector(q0, q1, 0, 1, 2, 3, 4, 5, 6, 7));
}

// Split 16 fp32 into packed bf16 hi/lo dword pairs and store as 4x b128.
__device__ __forceinline__ void pack_store(const float* r,
                                           unsigned short* ph, unsigned short* pl) {
    unsigned int h[8], l[8];
#pragma unroll
    for (int q = 0; q < 8; ++q) {
        const float x0 = r[2 * q], x1 = r[2 * q + 1];
        const unsigned int u0 = __float_as_uint(x0);
        const unsigned int u1 = __float_as_uint(x1);
        h[q] = (u0 >> 16) | (u1 & 0xFFFF0000u);
        const float f0 = __uint_as_float(u0 & 0xFFFF0000u);
        const float f1 = __uint_as_float(u1 & 0xFFFF0000u);
        l[q] = (__float_as_uint(x0 - f0) >> 16) |
               (__float_as_uint(x1 - f1) & 0xFFFF0000u);
    }
    u32x4 H0 = {h[0], h[1], h[2], h[3]}, H1 = {h[4], h[5], h[6], h[7]};
    u32x4 L0 = {l[0], l[1], l[2], l[3]}, L1 = {l[4], l[5], l[6], l[7]};
    *(u32x4*)(ph)     = H0;
    *(u32x4*)(ph + 8) = H1;
    *(u32x4*)(pl)     = L0;
    *(u32x4*)(pl + 8) = L1;
}

// C = A[M,K] * W[N,K]^T   (both row-major, K contiguous)
// If emb != nullptr:  C[m,n] += alpha * emb[inds[b, stage+nseq], n]   (proj epilogue)
__global__ void __launch_bounds__(256)
gemm_bf16x3(const float* __restrict__ A, const float* __restrict__ W,
            float* __restrict__ C, int K, int N,
            const float* __restrict__ emb, const int* __restrict__ inds,
            int stage, float alpha)
{
    __shared__ unsigned short sAh[BM * LDT], sAl[BM * LDT];
    __shared__ unsigned short sBh[BN * LDT], sBl[BN * LDT];

    const int tid  = threadIdx.x;
    const int bx   = blockIdx.x;            // N tile
    const int by   = blockIdx.y;            // M tile
    const int lane = tid & 31;
    const int wave = tid >> 5;
    const int wm   = (wave >> 2) * 64;      // wave 64x32 patch inside 128x128 tile
    const int wn   = (wave & 3) * 32;

    // cooperative loader mapping: 256 thr -> 128 rows x 2 half-rows of 16 floats
    const int lr = tid >> 1;
    const int lc = (tid & 1) * 16;
    const float* Ag = A + (size_t)(by * BM + lr) * K + lc;
    const float* Wg = W + (size_t)(bx * BN + lr) * K + lc;
    unsigned short* pah = sAh + lr * LDT + lc;
    unsigned short* pal = sAl + lr * LDT + lc;
    unsigned short* pbh = sBh + lr * LDT + lc;
    unsigned short* pbl = sBl + lr * LDT + lc;

    v8f acc[4][2];
#pragma unroll
    for (int mi = 0; mi < 4; ++mi)
#pragma unroll
        for (int ni = 0; ni < 2; ++ni)
#pragma unroll
            for (int v = 0; v < 8; ++v) acc[mi][ni][v] = 0.0f;

    float ra[16], rb[16];
#pragma unroll
    for (int j = 0; j < 16; j += 4) {       // prologue: first K-slab
        *(float4*)(ra + j) = *(const float4*)(Ag + j);
        *(float4*)(rb + j) = *(const float4*)(Wg + j);
    }

    for (int kb = 0; kb < K; kb += BK) {
        pack_store(ra, pah, pal);
        pack_store(rb, pbh, pbl);
        __syncthreads();

        if (kb + BK < K) {                  // pipeline next slab under the WMMAs
#pragma unroll
            for (int j = 0; j < 16; j += 4) {
                *(float4*)(ra + j) = *(const float4*)(Ag + kb + BK + j);
                *(float4*)(rb + j) = *(const float4*)(Wg + kb + BK + j);
            }
        }

        bf16x16 ah[4], al[4], bh[2], bl[2];
#pragma unroll
        for (int mi = 0; mi < 4; ++mi) {
            ah[mi] = fragA(sAh, lane, wm + mi * 16);
            al[mi] = fragA(sAl, lane, wm + mi * 16);
        }
#pragma unroll
        for (int ni = 0; ni < 2; ++ni) {
            bh[ni] = fragB(sBh, lane, wn + ni * 16);
            bl[ni] = fragB(sBl, lane, wn + ni * 16);
        }
#pragma unroll
        for (int mi = 0; mi < 4; ++mi)
#pragma unroll
            for (int ni = 0; ni < 2; ++ni) {
                acc[mi][ni] = wmma_bf16(ah[mi], bh[ni], acc[mi][ni]);  // hi*hi
                acc[mi][ni] = wmma_bf16(ah[mi], bl[ni], acc[mi][ni]);  // hi*lo
                acc[mi][ni] = wmma_bf16(al[mi], bh[ni], acc[mi][ni]);  // lo*hi
            }
        __syncthreads();
    }

    // epilogue: C layout is VGPRv <-> M = v + 8*(lane>=16), N = lane&15
    const int cn = lane & 15;
    const int cm = (lane >> 4) * 8;
#pragma unroll
    for (int mi = 0; mi < 4; ++mi)
#pragma unroll
        for (int ni = 0; ni < 2; ++ni) {
            const int gm0 = by * BM + wm + mi * 16 + cm;
            const int gn  = bx * BN + wn + ni * 16 + cn;
#pragma unroll
            for (int v = 0; v < 8; ++v) {
                const int gm = gm0 + v;
                float val = acc[mi][ni][v];
                if (emb != nullptr) {
                    const int b    = gm >> 9;          // row -> batch
                    const int nseq = gm & (SEQ_N - 1); // row -> position
                    const int row  = inds[b * INDS_LD + stage + nseq];
                    val += alpha * emb[(size_t)row * INNER + gn];
                }
                C[(size_t)gm * N + gn] = val;
            }
        }
}

// RMS-norm (no mean) + scale/shift + exact GELU, one block (256 thr) per row of 1024
__global__ void __launch_bounds__(256)
norm_gelu(const float* __restrict__ x, float* __restrict__ y,
          const float* __restrict__ w, const float* __restrict__ b)
{
    __shared__ float red[256];
    const int row = blockIdx.x;
    const int tid = threadIdx.x;
    const float* xr = x + (size_t)row * INNER;

    float4 xv = *(const float4*)(xr + tid * 4);
    red[tid] = xv.x * xv.x + xv.y * xv.y + xv.z * xv.z + xv.w * xv.w;
    __syncthreads();
#pragma unroll
    for (int s = 128; s > 0; s >>= 1) {
        if (tid < s) red[tid] += red[tid + s];
        __syncthreads();
    }
    const float rms = rsqrtf(red[0] * (1.0f / (float)INNER) + 1e-6f);

    float e[4] = {xv.x, xv.y, xv.z, xv.w};
    float o[4];
#pragma unroll
    for (int j = 0; j < 4; ++j) {
        const int c = tid * 4 + j;
        float t = e[j] * rms * w[c] + b[c];
        o[j] = 0.5f * t * (1.0f + erff(t * 0.70710678118654752f));
    }
    float4 ov = {o[0], o[1], o[2], o[3]};
    *(float4*)(y + (size_t)row * INNER + tid * 4) = ov;
}

extern "C" void kernel_launch(void* const* d_in, const int* in_sizes, int n_in,
                              void* d_out, int out_size, void* d_ws, size_t ws_size,
                              hipStream_t stream)
{
    const float* state = (const float*)d_in[0];   // [2,512,4096]
    const int*   inds  = (const int*)  d_in[1];   // [2,515]
    const float* emb_w = (const float*)d_in[2];   // [3,32000,1024]
    const float* proj0 = (const float*)d_in[3];   // [1024,4096]
    const float* projw = (const float*)d_in[4];   // [2,1024,1024]
    const float* headw = (const float*)d_in[5];   // [3,32000,1024]
    const float* lnw   = (const float*)d_in[6];   // [3,1024]
    const float* lnb   = (const float*)d_in[7];   // [3,1024]
    float* out = (float*)d_out;                   // [3,2,512,32000]

    float* tmp  = (float*)d_ws;                   // [1024,1024] pre-norm state
    float* sbuf = tmp + (size_t)1024 * 1024;      // [1024,1024] activated state

    const double sw    = pow(0.5, 0.5 / 3.0);     // STATE_W
    const float  alpha = (float)(sqrt((1.0 - sw * sw) * (INNER / 2.0)) / sw);

    for (int i = 0; i < 3; ++i) {
        const float* Ap = (i == 0) ? state : sbuf;
        const float* Wp = (i == 0) ? proj0 : (projw + (size_t)(i - 1) * INNER * INNER);
        const int    K  = (i == 0) ? 4096 : INNER;

        // proj GEMM (M=1024, N=1024) + fused alpha*emb gather
        gemm_bf16x3<<<dim3(INNER / BN, 1024 / BM), 256, 0, stream>>>(
            Ap, Wp, tmp, K, INNER,
            emb_w + (size_t)i * VOCAB * INNER, inds, i, alpha);

        // RMS norm + scale/shift + exact GELU
        norm_gelu<<<dim3(1024), 256, 0, stream>>>(
            tmp, sbuf, lnw + i * INNER, lnb + i * INNER);

        // head GEMM (M=1024, N=32000) straight to output
        gemm_bf16x3<<<dim3(VOCAB / BN, 1024 / BM), 256, 0, stream>>>(
            sbuf, headw + (size_t)i * VOCAB * INNER,
            out + (size_t)i * 1024 * VOCAB, INNER, VOCAB,
            nullptr, nullptr, 0, 0.0f);
    }
}
